// BiLSTM_CRF_17394617548940
// MI455X (gfx1250) — compile-verified, matrix operand order checked by reading, not run on hardware
//
#include <hip/hip_runtime.h>
#include <hip/hip_bf16.h>

// ---------------------------------------------------------------------------
// BiLSTM-CRF for MI455X (gfx1250), wave32, bf16 WMMA (v_wmma_f32_16x16x32_bf16)
// + CDNA5 async global->LDS staging (GLOBAL_LOAD_ASYNC_TO_LDS_B128 / ASYNCcnt)
//
// Pipeline (all on `stream`):
//   1) init_convert_kernel : emb -> bf16, Wcat[d] = [Wih_d | Whh_d] bf16,
//                            zero h ping-pong buffers + barrier counters
//   2) bilstm_kernel       : persistent kernel, 32 WGs (16 per direction),
//                            each WG owns a 32-wide slice of H. Per step:
//                            async-stage A panel [emb_row | h] (64x768 bf16)
//                            into LDS, fused GEMM vs Wcat^T (K=768) via WMMA,
//                            f32 accum, activations in LDS, h slice -> global,
//                            device-scope spin barrier, ping-pong h buffers.
//   3) emissions_kernel    : em[b,s,t] = blin[t] + <lstm_out[s,b,:], Wlin[t,:]>
//   4) crf_kernel          : gold score + forward logsumexp per batch lane,
//                            mean(log_z - num) -> d_out[0]
// ---------------------------------------------------------------------------

typedef __attribute__((ext_vector_type(16))) __bf16 v16bf;
typedef __attribute__((ext_vector_type(8)))  __bf16 v8bf;
typedef __attribute__((ext_vector_type(8)))  float  v8f;

#define V_SZ 10000
#define E_SZ 256
#define H_SZ 512
#define B_SZ 64
#define S_SZ 512
#define G_SZ 2048          // 4*H
#define K_SZ 768           // E + H  (fused K)
#define K_PAD 776          // +8 bf16 pad: row stride 388 dwords == 4 mod 64 banks
#define T_SZ 5
#define NWG  16            // workgroups per direction

// workspace layout (bytes, all 16B aligned)
#define OFF_EMB   ((size_t)0)
#define OFF_WCAT  (OFF_EMB  + (size_t)V_SZ * E_SZ * 2)            // emb bf16
#define OFF_HBUF  (OFF_WCAT + (size_t)2 * G_SZ * K_SZ * 2)        // Wcat bf16
#define OFF_LSTM  (OFF_HBUF + (size_t)2 * 2 * B_SZ * H_SZ * 2)    // h ping-pong bf16
#define OFF_EMIS  (OFF_LSTM + (size_t)S_SZ * B_SZ * 2 * H_SZ * 2) // lstm_out bf16
#define OFF_BAR   (OFF_EMIS + (size_t)B_SZ * S_SZ * T_SZ * 4)     // emissions f32

__device__ __forceinline__ float sigmoidf_(float x) {
  return 1.0f / (1.0f + __expf(-x));
}
__device__ __forceinline__ float tanhf_(float x) {
  x = fminf(fmaxf(x, -15.0f), 15.0f);
  float e = __expf(2.0f * x);
  return (e - 1.0f) / (e + 1.0f);
}

// CDNA5 async copy: 16 bytes global -> LDS, tracked by ASYNCcnt.
// vdst = LDS byte offset (low 32 bits of flat shared address), vaddr = 64-bit VA.
__device__ __forceinline__ void async_copy16(unsigned lds_off, const void* gptr) {
  asm volatile("global_load_async_to_lds_b128 %0, %1, off"
               :: "v"(lds_off), "v"((unsigned long long)(uintptr_t)gptr)
               : "memory");
}
__device__ __forceinline__ void wait_async0() {
  asm volatile("s_wait_asynccnt 0" ::: "memory");
}

// ---------------------------------------------------------------------------
// 1) conversions + state init (re-run every call for determinism)
// ---------------------------------------------------------------------------
__global__ void init_convert_kernel(const float* __restrict__ emb,
                                    const float* __restrict__ Wih_f,
                                    const float* __restrict__ Whh_f,
                                    const float* __restrict__ Wih_b,
                                    const float* __restrict__ Whh_b,
                                    char* __restrict__ ws) {
  __bf16* emb_bf = (__bf16*)(ws + OFF_EMB);
  __bf16* Wcat   = (__bf16*)(ws + OFF_WCAT);
  __bf16* hbuf   = (__bf16*)(ws + OFF_HBUF);
  int*    bar    = (int*)(ws + OFF_BAR);

  size_t tid    = (size_t)blockIdx.x * blockDim.x + threadIdx.x;
  size_t stride = (size_t)gridDim.x * blockDim.x;

  for (size_t i = tid; i < (size_t)V_SZ * E_SZ; i += stride)
    emb_bf[i] = (__bf16)emb[i];

  // Wcat[d][g][k] : k<E -> Wih_d[g][k], else Whh_d[g][k-E]   (row-major [g][K])
  for (size_t i = tid; i < (size_t)2 * G_SZ * K_SZ; i += stride) {
    int d = (int)(i / ((size_t)G_SZ * K_SZ));
    int r = (int)(i % ((size_t)G_SZ * K_SZ));
    int g = r / K_SZ, k = r % K_SZ;
    float v;
    if (k < E_SZ) v = (d ? Wih_b : Wih_f)[g * E_SZ + k];
    else          v = (d ? Whh_b : Whh_f)[g * H_SZ + (k - E_SZ)];
    Wcat[i] = (__bf16)v;
  }

  for (size_t i = tid; i < (size_t)2 * 2 * B_SZ * H_SZ; i += stride)
    hbuf[i] = (__bf16)0.0f;

  if (tid == 0) { bar[0] = 0; bar[1] = 0; }
}

// ---------------------------------------------------------------------------
// 2) persistent BiLSTM recurrence, async LDS staging, bf16 WMMA, f32 accum
//    grid = 32 blocks (block 0..15 fwd, 16..31 bwd), 256 threads = 8 waves
// ---------------------------------------------------------------------------
__global__ __launch_bounds__(256, 1)
void bilstm_kernel(const int* __restrict__ x,
                   const float* __restrict__ b_f,
                   const float* __restrict__ b_b,
                   char* __restrict__ ws) {
  const __bf16* emb_bf   = (const __bf16*)(ws + OFF_EMB);
  const __bf16* Wcat     = (const __bf16*)(ws + OFF_WCAT);
  __bf16*       hbuf     = (__bf16*)(ws + OFF_HBUF);
  __bf16*       lstm_out = (__bf16*)(ws + OFF_LSTM);
  int*          bar      = (int*)(ws + OFF_BAR);

  const int dir = blockIdx.x >> 4;      // 0 = forward, 1 = backward
  const int wg  = blockIdx.x & 15;      // slice id within direction
  const int j0  = wg * 32;              // owned hidden columns [j0, j0+32)

  __shared__ __bf16 apanel[B_SZ][K_PAD]; // A panel: [emb_row | h] per batch row
  __shared__ float  gates[4][B_SZ][32];  // i,f,g,o for the owned slice
  __shared__ float  cbuf[B_SZ][32];      // persistent cell state (f32)
  __shared__ float  hown[B_SZ][32];      // persistent own h slice (f32)

  const int tid = threadIdx.x;
  for (int e = tid; e < B_SZ * 32; e += 256) {
    cbuf[e >> 5][e & 31] = 0.0f;
    hown[e >> 5][e & 31] = 0.0f;
  }

  const int wave = tid >> 5;
  const int lane = tid & 31;
  const int l16  = lane & 15;
  const int hi   = lane >> 4;                 // lane half (ISA K-run select)
  const int gate = wave >> 1;                 // 0..3 -> i,f,g,o
  const int csub = (wave & 1) * 16 + l16;     // column within 32-wide slice
  const int gcol = gate * H_SZ + j0 + csub;   // global gate column (N index)
  const float biasv = (dir ? b_b : b_f)[gcol];
  const __bf16* Wrow = Wcat + ((size_t)dir * G_SZ + gcol) * K_SZ; // B operand row
  const int koff = hi * 8;                    // per-lane K sub-offset

  __bf16* hb_dir = hbuf + (size_t)dir * 2 * B_SZ * H_SZ;
  const unsigned apbase = (unsigned)(uintptr_t)(&apanel[0][0]);

  __syncthreads();

  for (int it = 0; it < S_SZ; ++it) {
    const int s  = dir ? (S_SZ - 1 - it) : it;
    const int rp = it & 1;
    const __bf16* hrd = hb_dir + (size_t)rp * B_SZ * H_SZ;
    __bf16*       hwr = hb_dir + (size_t)(rp ^ 1) * B_SZ * H_SZ;

    // ---- async-stage A panel into LDS (ASYNCcnt) ------------------------
    // embedding gather: 64 rows x 256 bf16 = 2048 x 16B chunks
    for (int c = tid; c < 2048; c += 256) {
      int b = c >> 5, q = c & 31;
      const __bf16* g = emb_bf + (size_t)x[b * S_SZ + s] * E_SZ + q * 8;
      async_copy16(apbase + (unsigned)(b * (K_PAD * 2) + q * 16), g);
    }
    // recurrent h: 64 rows x 512 bf16 = 4096 x 16B chunks
    for (int c = tid; c < 4096; c += 256) {
      int b = c >> 6, q = c & 63;
      const __bf16* g = hrd + (size_t)b * H_SZ + q * 8;
      async_copy16(apbase + (unsigned)(b * (K_PAD * 2) + E_SZ * 2 + q * 16), g);
    }
    wait_async0();
    __syncthreads();

    // ---- fused GEMM: gates = [emb|h] x Wcat^T + b, K = 768 --------------
    v8f acc[4];
    #pragma unroll
    for (int mt = 0; mt < 4; ++mt)
      #pragma unroll
      for (int r = 0; r < 8; ++r) acc[mt][r] = biasv;   // bias pre-loaded in C

    #pragma unroll 4
    for (int kt = 0; kt < 24; ++kt) {
      const int k0 = kt * 32;
      union { v16bf v; v8bf h[2]; } bfrag;
      bfrag.h[0] = *(const v8bf*)(Wrow + k0 + koff);
      bfrag.h[1] = *(const v8bf*)(Wrow + k0 + koff + 16);
      #pragma unroll
      for (int mt = 0; mt < 4; ++mt) {
        const __bf16* ap = &apanel[mt * 16 + l16][k0 + koff];
        union { v16bf v; v8bf h[2]; } afrag;
        afrag.h[0] = *(const v8bf*)ap;         // ds_load_b128
        afrag.h[1] = *(const v8bf*)(ap + 16);  // ds_load_b128
        acc[mt] = __builtin_amdgcn_wmma_f32_16x16x32_bf16(
            false, afrag.v, false, bfrag.v, (short)0, acc[mt], false, false);
      }
    }

    // scatter gate tiles to LDS (C/D layout: VGPR r -> M = r + 8*hi)
    #pragma unroll
    for (int mt = 0; mt < 4; ++mt)
      #pragma unroll
      for (int r = 0; r < 8; ++r)
        gates[gate][mt * 16 + hi * 8 + r][csub] = acc[mt][r];
    __syncthreads();

    // ---- element-wise LSTM cell update over the 64x32 owned slice -------
    #pragma unroll
    for (int e = 0; e < 8; ++e) {
      int idx = tid + e * 256;
      int b = idx >> 5, jj = idx & 31;
      float gi = gates[0][b][jj], gf = gates[1][b][jj];
      float gg = gates[2][b][jj], go = gates[3][b][jj];
      float co = cbuf[b][jj], ho = hown[b][jj];
      float m  = (x[b * S_SZ + s] != 0) ? 1.0f : 0.0f;
      float cn = sigmoidf_(gf) * co + sigmoidf_(gi) * tanhf_(gg);
      float hn = sigmoidf_(go) * tanhf_(cn);
      float cu = m * cn + (1.0f - m) * co;
      float hu = m * hn + (1.0f - m) * ho;
      cbuf[b][jj] = cu;
      hown[b][jj] = hu;
      hwr[(size_t)b * H_SZ + j0 + jj] = (__bf16)hu;                  // recurrent state
      lstm_out[((size_t)s * B_SZ + b) * (2 * H_SZ) + dir * H_SZ + j0 + jj] =
          (__bf16)(hn * m);                                          // hs = h * m
    }

    // ---- device-scope step barrier across the 16 WGs of this direction --
    __threadfence();
    __syncthreads();
    if (tid == 0) {
      __hip_atomic_fetch_add(&bar[dir], 1, __ATOMIC_ACQ_REL, __HIP_MEMORY_SCOPE_AGENT);
      const int target = NWG * (it + 1);
      while (__hip_atomic_load(&bar[dir], __ATOMIC_ACQUIRE, __HIP_MEMORY_SCOPE_AGENT) < target)
        __builtin_amdgcn_s_sleep(2);
    }
    __syncthreads();
  }
}

// ---------------------------------------------------------------------------
// 3) emissions: em[b,s,t] = blin[t] + dot(lstm_out[s,b,:], Wlin[t,:])
// ---------------------------------------------------------------------------
__global__ void emissions_kernel(const float* __restrict__ Wlin,
                                 const float* __restrict__ blin,
                                 char* __restrict__ ws) {
  const __bf16* lstm_out = (const __bf16*)(ws + OFF_LSTM);
  float*        em       = (float*)(ws + OFF_EMIS);
  int idx = blockIdx.x * blockDim.x + threadIdx.x;
  if (idx >= B_SZ * S_SZ * T_SZ) return;
  int t = idx % T_SZ;
  int s = (idx / T_SZ) % S_SZ;
  int b = idx / (T_SZ * S_SZ);
  const __bf16* hrow = lstm_out + ((size_t)s * B_SZ + b) * (2 * H_SZ);
  const float*  wrow = Wlin + t * (2 * H_SZ);
  float sum = blin[t];
  #pragma unroll 8
  for (int k = 0; k < 2 * H_SZ; ++k) sum += (float)hrow[k] * wrow[k];
  em[idx] = sum;   // layout [b][s][t]
}

// ---------------------------------------------------------------------------
// 4) CRF: gold score + forward algorithm, one batch element per lane
// ---------------------------------------------------------------------------
__global__ void crf_kernel(const int* __restrict__ tags,
                           const int* __restrict__ lengths,
                           const float* __restrict__ trans,
                           const float* __restrict__ start_t,
                           const float* __restrict__ end_t,
                           const char* __restrict__ ws,
                           float* __restrict__ out) {
  const float* em = (const float*)(ws + OFF_EMIS);
  __shared__ float tr[T_SZ * T_SZ];
  __shared__ float red[B_SZ];
  const int tid = threadIdx.x;
  if (tid < T_SZ * T_SZ) tr[tid] = trans[tid];
  __syncthreads();

  const int b = tid;
  const int*   tg = tags + b * S_SZ;
  const float* eb = em + (size_t)b * S_SZ * T_SZ;

  float a[T_SZ];
  int t0 = tg[0];
  float num = start_t[t0] + eb[t0];
  #pragma unroll
  for (int t = 0; t < T_SZ; ++t) a[t] = start_t[t] + eb[t];
  int tprev = t0;

  for (int s = 1; s < S_SZ; ++s) {
    int tc = tg[s];
    if (tc != 4) {                       // PAD_TAG == 4, pads are trailing
      num += tr[tprev * T_SZ + tc] + eb[s * T_SZ + tc];
      tprev = tc;
      float an[T_SZ];
      #pragma unroll
      for (int t2 = 0; t2 < T_SZ; ++t2) {
        float mx = -3.4e38f;
        #pragma unroll
        for (int t1 = 0; t1 < T_SZ; ++t1)
          mx = fmaxf(mx, a[t1] + tr[t1 * T_SZ + t2]);
        float ssum = 0.0f;
        #pragma unroll
        for (int t1 = 0; t1 < T_SZ; ++t1)
          ssum += __expf(a[t1] + tr[t1 * T_SZ + t2] - mx);
        an[t2] = mx + __logf(ssum) + eb[s * T_SZ + t2];
      }
      #pragma unroll
      for (int t = 0; t < T_SZ; ++t) a[t] = an[t];
    }
  }

  int len = lengths[b];
  num += end_t[tg[len - 1]];

  float mx = -3.4e38f;
  #pragma unroll
  for (int t = 0; t < T_SZ; ++t) mx = fmaxf(mx, a[t] + end_t[t]);
  float ssum = 0.0f;
  #pragma unroll
  for (int t = 0; t < T_SZ; ++t) ssum += __expf(a[t] + end_t[t] - mx);
  float log_z = mx + __logf(ssum);

  red[b] = log_z - num;
  __syncthreads();
  if (b == 0) {
    float acc = 0.0f;
    for (int i = 0; i < B_SZ; ++i) acc += red[i];
    out[0] = acc / (float)B_SZ;
  }
}

// ---------------------------------------------------------------------------
extern "C" void kernel_launch(void* const* d_in, const int* in_sizes, int n_in,
                              void* d_out, int out_size, void* d_ws, size_t ws_size,
                              hipStream_t stream) {
  const int*   x       = (const int*)d_in[0];
  const int*   tags    = (const int*)d_in[1];
  const int*   lengths = (const int*)d_in[2];
  const float* emb     = (const float*)d_in[3];
  const float* Wih_f   = (const float*)d_in[4];
  const float* Whh_f   = (const float*)d_in[5];
  const float* b_f     = (const float*)d_in[6];
  const float* Wih_b   = (const float*)d_in[7];
  const float* Whh_b   = (const float*)d_in[8];
  const float* b_b     = (const float*)d_in[9];
  const float* Wlin    = (const float*)d_in[10];
  const float* blin    = (const float*)d_in[11];
  const float* trans   = (const float*)d_in[12];
  const float* start_t = (const float*)d_in[13];
  const float* end_t   = (const float*)d_in[14];
  char*  ws  = (char*)d_ws;
  float* out = (float*)d_out;
  (void)in_sizes; (void)n_in; (void)out_size; (void)ws_size;

  init_convert_kernel<<<2048, 256, 0, stream>>>(emb, Wih_f, Whh_f, Wih_b, Whh_b, ws);
  bilstm_kernel<<<2 * NWG, 256, 0, stream>>>(x, b_f, b_b, ws);
  emissions_kernel<<<(B_SZ * S_SZ * T_SZ + 255) / 256, 256, 0, stream>>>(Wlin, blin, ws);
  crf_kernel<<<1, B_SZ, 0, stream>>>(tags, lengths, trans, start_t, end_t, ws, out);
}
